// MOEModel_16896401343014
// MI455X (gfx1250) — matrix operand order repaired and checked by reading, not compiled, not run on hardware
//
#include <hip/hip_runtime.h>
#include <hip/hip_bf16.h>
#include <cmath>

// ---------------------------------------------------------------------------
// MOE multimodal model for MI455X (gfx1250), wave32 + WMMA f32_16x16x32_f16.
//
// Exact-math simplification: g1/g2 in the reference are
//   softmax(.., -1).mean(-1) == 1/N == 0.1 for every token, so the whole
//   cross-attention / LN / MLP gating block is dead code; the 0.1 gates on
//   Q and K fold into the modulated-attention score scale (0.01/sqrt(512)).
//
// MoE (dominant ~97/130 GFLOP): operands pre-converted to f16 once, then the
// expert GEMM stages tiles via GLOBAL_LOAD_ASYNC_TO_LDS_B128 (ASYNCcnt) with
// double-buffered LDS, one barrier per K-step. Other GEMMs: fp32->f16 via
// register-staged double buffer, one barrier per K-step.
//
// All GEMM dims are multiples of 64 (M,N) and 32 (K): no edge guards.
// ---------------------------------------------------------------------------

typedef float     v8f  __attribute__((ext_vector_type(8)));
typedef _Float16  v8h  __attribute__((ext_vector_type(8)));
typedef _Float16  v16h __attribute__((ext_vector_type(16)));
typedef long long i64;

#define DB   16
#define DSI  256
#define DST  512
#define DE   768
#define DH   512
#define DN   10
#define DNH  8
#define DHD  64
#define TILE 64
#define KST  32
#define LDT  40   // LDS tile row stride in halves (padded vs bank conflicts)

// ---- WMMA fragment loads from LDS tiles laid out [row][k] (stride LDT) ----
// A fragment (16x32 MxK): lane row = lane&15; halves 0..7 -> K=koff+j,
// halves 8..15 -> K=16+koff+j, koff=(lane>>4)*8  (ISA 7.12.2).
__device__ __forceinline__ v16h loadAFrag(const _Float16* tile, int rowOff, int lane) {
  int row  = rowOff + (lane & 15);
  int koff = (lane >> 4) * 8;
  const v8h* p0 = (const v8h*)(tile + row * LDT + koff);
  const v8h* p1 = (const v8h*)(tile + row * LDT + 16 + koff);
  v8h lo = *p0, hi = *p1;
  v16h r;
#pragma unroll
  for (int i = 0; i < 8; ++i) { r[i] = lo[i]; r[8 + i] = hi[i]; }
  return r;
}
// B fragment (32x16 KxN): lane col = lane&15; halves 0..7 -> K=kb+j, halves 8..15 -> K=16+kb+j, kb=(lane>>4)*8.
__device__ __forceinline__ v16h loadBFrag(const _Float16* tile, int colOff, int lane) {
  int col = colOff + (lane & 15);
  int kb  = (lane >> 4) * 8;
  const v8h* p0 = (const v8h*)(tile + col * LDT + kb);
  const v8h* p1 = (const v8h*)(tile + col * LDT + kb + 16);
  v8h lo = *p0, hi = *p1;
  v16h r;
#pragma unroll
  for (int i = 0; i < 8; ++i) { r[i] = lo[i]; r[8 + i] = hi[i]; }
  return r;
}

__device__ __forceinline__ void cvtStore8(_Float16* dst, float4 q0, float4 q1) {
  dst[0] = (_Float16)q0.x; dst[1] = (_Float16)q0.y;
  dst[2] = (_Float16)q0.z; dst[3] = (_Float16)q0.w;
  dst[4] = (_Float16)q1.x; dst[5] = (_Float16)q1.y;
  dst[6] = (_Float16)q1.z; dst[7] = (_Float16)q1.w;
}

// ---- CDNA5 async global->LDS (16B per lane, tracked by ASYNCcnt) ----------
// LDS operand = low 32 bits of the flat pointer.
__device__ __forceinline__ void asyncCopy16(void* lds, const void* gsrc) {
  unsigned l = (unsigned)(unsigned long long)lds;
  asm volatile("global_load_async_to_lds_b128 %0, %1, off"
               :: "v"(l), "v"((unsigned long long)gsrc) : "memory");
}
__device__ __forceinline__ void asyncWait0() {
  asm volatile("s_wait_asynccnt 0x0" ::: "memory");
}

// ---------------------------------------------------------------------------
// fp32 -> f16 conversion (one-time, for MoE operands)
// ---------------------------------------------------------------------------
__global__ __launch_bounds__(256) void k_cvt_f16(const float* __restrict__ in,
                                                 _Float16* __restrict__ out, i64 n) {
  i64 i = (i64)blockIdx.x * 256 + threadIdx.x;
  i64 stride = (i64)gridDim.x * 256;
  for (; i < n; i += stride) out[i] = (_Float16)in[i];
}

// ---------------------------------------------------------------------------
// Gating: one wave per token, fp32 dot products (top-2 pick is discrete).
// ---------------------------------------------------------------------------
__global__ __launch_bounds__(256) void k_gating(const float* __restrict__ X,
                                                const float* __restrict__ GW,
                                                const float* __restrict__ GB,
                                                float* __restrict__ Wt, int M) {
  int wave = threadIdx.x >> 5, lane = threadIdx.x & 31;
  int tok = blockIdx.x * 8 + wave;
  if (tok >= M) return;
  const int NCH = DE / 32;
  float xv[NCH];
  const float* xr = X + (i64)tok * DE;
#pragma unroll
  for (int j = 0; j < NCH; ++j) xv[j] = xr[lane + 32 * j];
  float logit[DN];
#pragma unroll
  for (int n = 0; n < DN; ++n) {
    const float* g = GW + (i64)n * DE;
    float s = 0.f;
#pragma unroll
    for (int j = 0; j < NCH; ++j) s += xv[j] * g[lane + 32 * j];
#pragma unroll
    for (int o = 16; o > 0; o >>= 1) s += __shfl_xor(s, o, 32);
    logit[n] = s + GB[n];
  }
  if (lane == 0) {
    float mx = logit[0];
#pragma unroll
    for (int n = 1; n < DN; ++n) mx = fmaxf(mx, logit[n]);
    float p[DN], s = 0.f;
#pragma unroll
    for (int n = 0; n < DN; ++n) { p[n] = expf(logit[n] - mx); s += p[n]; }
    float inv = 1.f / s;
#pragma unroll
    for (int n = 0; n < DN; ++n) p[n] *= inv;
    int i0 = 0;
#pragma unroll
    for (int n = 1; n < DN; ++n) if (p[n] > p[i0]) i0 = n;  // first max = top_k tie rule
    int i1 = (i0 == 0) ? 1 : 0;
#pragma unroll
    for (int n = 0; n < DN; ++n) if (n != i0 && n != i1 && p[n] > p[i1]) i1 = n;
    float* o = Wt + (i64)tok * DN;
#pragma unroll
    for (int n = 0; n < DN; ++n) o[n] = (n == i0) ? p[i0] : (n == i1) ? p[i1] : 0.f;
  }
}

// ---------------------------------------------------------------------------
// Fused MoE: OUT[m,h] = sum_n Wt[m,n] * (X[m,:] . EW[n,h,:] + EB[n,h]).
// f16 operands, async global->LDS staging, double-buffered, 1 barrier/step.
// Loop flattened over (expert, k-step) so the pipeline never drains.
// ---------------------------------------------------------------------------
__global__ __launch_bounds__(256) void k_moe(const _Float16* __restrict__ Xh,
                                             const _Float16* __restrict__ EWh,
                                             const float* __restrict__ EB,
                                             const float* __restrict__ Wt,
                                             float* __restrict__ OUT) {
  __shared__ _Float16 As[2][TILE][LDT];
  __shared__ _Float16 Bs[2][TILE][LDT];
  int rowBase = blockIdx.y * TILE;
  int colBase = blockIdx.x * TILE;
  int t = threadIdx.x, wave = t >> 5, lane = t & 31;
  int rt = wave >> 1;            // row subtile 0..3
  int ct = (wave & 1) * 2;       // col subtiles {0,1} or {2,3}
  int lr = (t * 8) >> 5, lc = (t * 8) & 31;  // this thread's 16B staging slot

  const _Float16* Xrow = Xh + (i64)(rowBase + lr) * DE + lc;
  const _Float16* Wrow = EWh + (i64)(colBase + lr) * DE + lc;  // + n*DH*DE per expert
  const int NKT = DE / KST;        // 24
  const int STEPS = DN * NKT;      // 240

  auto prefetch = [&](int s, int buf) {
    int n = s / NKT, kt = s - n * NKT;
    asyncCopy16(&As[buf][lr][lc], Xrow + kt * KST);
    asyncCopy16(&Bs[buf][lr][lc], Wrow + (i64)n * DH * DE + kt * KST);
  };

  v8f comb0 = {}, comb1 = {};
  v8f acc0 = {}, acc1 = {};
  prefetch(0, 0);
  for (int s = 0; s < STEPS; ++s) {
    int buf = s & 1;
    asyncWait0();        // my copies for step s have landed in LDS
    __syncthreads();     // everyone's copies visible; other buffer free
    if (s + 1 < STEPS) prefetch(s + 1, buf ^ 1);
    int n = s / NKT, kt = s - n * NKT;
    if (kt == 0) { acc0 = (v8f){}; acc1 = (v8f){}; }
    v16h af  = loadAFrag(&As[buf][0][0], rt * 16, lane);
    v16h bf0 = loadBFrag(&Bs[buf][0][0], ct * 16, lane);
    v16h bf1 = loadBFrag(&Bs[buf][0][0], (ct + 1) * 16, lane);
    acc0 = __builtin_amdgcn_wmma_f32_16x16x32_f16(false, af, false, bf0, (short)0, acc0, false, false);
    acc1 = __builtin_amdgcn_wmma_f32_16x16x32_f16(false, af, false, bf1, (short)0, acc1, false, false);
    if (kt == NKT - 1) {
      int rbase = rowBase + rt * 16 + (lane >> 4) * 8;
      int col0  = colBase + ct * 16 + (lane & 15);
      float e0 = EB[(i64)n * DH + col0];
      float e1 = EB[(i64)n * DH + col0 + 16];
#pragma unroll
      for (int v = 0; v < 8; ++v) {
        float w = Wt[(i64)(rbase + v) * DN + n];
        comb0[v] += w * (acc0[v] + e0);
        comb1[v] += w * (acc1[v] + e1);
      }
    }
  }
  int rbase = rowBase + rt * 16 + (lane >> 4) * 8;
  int col0  = colBase + ct * 16 + (lane & 15);
#pragma unroll
  for (int v = 0; v < 8; ++v) {
    OUT[(i64)(rbase + v) * DH + col0]      = comb0[v];
    OUT[(i64)(rbase + v) * DH + col0 + 16] = comb1[v];
  }
}

// ---------------------------------------------------------------------------
// Generic batched WMMA GEMM (fp32 in/out), register-staged double buffer.
//  TRANSB=true : C = s*(A[M,K] @ B[N,K]^T + bias)
//  TRANSB=false: C = s*(A[M,K] @ B[K,N]   + bias)
// Batch z decomposes as zo=z/innerDiv, zi=z%innerDiv with separate strides.
// ---------------------------------------------------------------------------
struct GemmP {
  const float* A; const float* B; const float* bias; float* C;
  int M, N, K, innerDiv;
  i64 lda, ldb, ldc;
  i64 aOuter, aInner, bOuter, bInner, cOuter, cInner;
  float outScale;
};

template <bool TRANSB>
__global__ __launch_bounds__(256) void k_gemm(GemmP g) {
  __shared__ _Float16 As[2][TILE][LDT];
  __shared__ _Float16 Bs[2][TILE][LDT];
  int z  = blockIdx.z;
  int zo = z / g.innerDiv, zi = z % g.innerDiv;
  const float* A = g.A + (i64)zo * g.aOuter + (i64)zi * g.aInner;
  const float* B = g.B + (i64)zo * g.bOuter + (i64)zi * g.bInner;
  float*       C = g.C + (i64)zo * g.cOuter + (i64)zi * g.cInner;
  int rowBase = blockIdx.y * TILE;
  int colBase = blockIdx.x * TILE;
  int t = threadIdx.x, wave = t >> 5, lane = t & 31;
  int rt = wave >> 1;
  int ct = (wave & 1) * 2;

  int ar = (t * 8) >> 5, ac = (t * 8) & 31;   // A staging slot
  int bk = (t * 8) >> 6, bn = (t * 8) & 63;   // B staging slot (NN path)

  float4 ra0, ra1, rb0, rb1;
  auto fetch = [&](int kt) {
    int k0 = kt * KST;
    const float4* a4 = (const float4*)(A + (i64)(rowBase + ar) * g.lda + k0 + ac);
    ra0 = a4[0]; ra1 = a4[1];
    if (TRANSB) {
      const float4* b4 = (const float4*)(B + (i64)(colBase + ar) * g.ldb + k0 + ac);
      rb0 = b4[0]; rb1 = b4[1];
    } else {
      const float4* b4 = (const float4*)(B + (i64)(k0 + bk) * g.ldb + colBase + bn);
      rb0 = b4[0]; rb1 = b4[1];
    }
  };
  auto stage = [&](int buf) {
    cvtStore8(&As[buf][ar][ac], ra0, ra1);
    if (TRANSB) {
      cvtStore8(&Bs[buf][ar][ac], rb0, rb1);
    } else {
      Bs[buf][bn + 0][bk] = (_Float16)rb0.x; Bs[buf][bn + 1][bk] = (_Float16)rb0.y;
      Bs[buf][bn + 2][bk] = (_Float16)rb0.z; Bs[buf][bn + 3][bk] = (_Float16)rb0.w;
      Bs[buf][bn + 4][bk] = (_Float16)rb1.x; Bs[buf][bn + 5][bk] = (_Float16)rb1.y;
      Bs[buf][bn + 6][bk] = (_Float16)rb1.z; Bs[buf][bn + 7][bk] = (_Float16)rb1.w;
    }
  };

  v8f acc0 = {}, acc1 = {};
  int nK = g.K / KST;
  fetch(0); stage(0);
  __syncthreads();
  for (int kt = 0; kt < nK; ++kt) {
    int buf = kt & 1;
    if (kt + 1 < nK) fetch(kt + 1);           // overlaps with compute below
    v16h af  = loadAFrag(&As[buf][0][0], rt * 16, lane);
    v16h bf0 = loadBFrag(&Bs[buf][0][0], ct * 16, lane);
    v16h bf1 = loadBFrag(&Bs[buf][0][0], (ct + 1) * 16, lane);
    acc0 = __builtin_amdgcn_wmma_f32_16x16x32_f16(false, af, false, bf0, (short)0, acc0, false, false);
    acc1 = __builtin_amdgcn_wmma_f32_16x16x32_f16(false, af, false, bf1, (short)0, acc1, false, false);
    if (kt + 1 < nK) stage(buf ^ 1);          // other buffer: free since last barrier
    __syncthreads();
  }
  int rbase = rowBase + rt * 16 + (lane >> 4) * 8;
  int col0  = colBase + ct * 16 + (lane & 15);
  int col1  = col0 + 16;
  float b0 = g.bias ? g.bias[col0] : 0.f;
  float b1 = g.bias ? g.bias[col1] : 0.f;
#pragma unroll
  for (int v = 0; v < 8; ++v) {
    C[(i64)(rbase + v) * g.ldc + col0] = g.outScale * (acc0[v] + b0);
    C[(i64)(rbase + v) * g.ldc + col1] = g.outScale * (acc1[v] + b1);
  }
}

// ---------------------------------------------------------------------------
// In-place row softmax, one block per row.
// ---------------------------------------------------------------------------
__global__ __launch_bounds__(256) void k_softmax(float* __restrict__ X, int cols) {
  __shared__ float red[8];
  i64 row = blockIdx.x;
  float* x = X + row * (i64)cols;
  int t = threadIdx.x, lane = t & 31, wv = t >> 5;
  float m = -1e30f;
  for (int c = t; c < cols; c += 256) m = fmaxf(m, x[c]);
#pragma unroll
  for (int o = 16; o > 0; o >>= 1) m = fmaxf(m, __shfl_xor(m, o, 32));
  if (lane == 0) red[wv] = m;
  __syncthreads();
  m = red[0];
#pragma unroll
  for (int i = 1; i < 8; ++i) m = fmaxf(m, red[i]);
  __syncthreads();
  float s = 0.f;
  for (int c = t; c < cols; c += 256) { float e = expf(x[c] - m); x[c] = e; s += e; }
#pragma unroll
  for (int o = 16; o > 0; o >>= 1) s += __shfl_xor(s, o, 32);
  if (lane == 0) red[wv] = s;
  __syncthreads();
  s = 0.f;
#pragma unroll
  for (int i = 0; i < 8; ++i) s += red[i];
  float inv = 1.f / s;
  for (int c = t; c < cols; c += 256) x[c] *= inv;
}

// ---------------------------------------------------------------------------
// fused = concat(fi, ft, axis=1).mean(axis=1); out = fused @ cls_w^T + cls_b.
// ---------------------------------------------------------------------------
__global__ __launch_bounds__(256) void k_classify(const float* __restrict__ FI,
                                                  const float* __restrict__ FT,
                                                  const float* __restrict__ CW,
                                                  const float* __restrict__ CB,
                                                  float* __restrict__ out) {
  __shared__ float fused[DH];
  int b = blockIdx.x, t = threadIdx.x;
  for (int c = t; c < DH; c += 256) {
    float s = 0.f;
    const float* fi = FI + (i64)b * DSI * DH + c;
    for (int r = 0; r < DSI; ++r) s += fi[(i64)r * DH];
    const float* ft = FT + (i64)b * DST * DH + c;
    for (int r = 0; r < DST; ++r) s += ft[(i64)r * DH];
    fused[c] = s * (1.f / (float)(DSI + DST));
  }
  __syncthreads();
  int w = t >> 5, lane = t & 31;
  if (w < 5) {
    float s = 0.f;
    for (int c = lane; c < DH; c += 32) s += fused[c] * CW[(i64)w * DH + c];
#pragma unroll
    for (int o = 16; o > 0; o >>= 1) s += __shfl_xor(s, o, 32);
    if (lane == 0) out[(i64)b * 5 + w] = s + CB[w];
  }
}

// ---------------------------------------------------------------------------
// Host launcher
// ---------------------------------------------------------------------------
extern "C" void kernel_launch(void* const* d_in, const int* in_sizes, int n_in,
                              void* d_out, int out_size, void* d_ws, size_t ws_size,
                              hipStream_t stream) {
  (void)in_sizes; (void)n_in; (void)out_size; (void)ws_size;
  const float* img_emb    = (const float*)d_in[0];
  const float* txt_emb    = (const float*)d_in[1];
  const float* gate_w_img = (const float*)d_in[2];
  const float* gate_b_img = (const float*)d_in[3];
  const float* exp_w_img  = (const float*)d_in[4];
  const float* exp_b_img  = (const float*)d_in[5];
  const float* gate_w_txt = (const float*)d_in[6];
  const float* gate_b_txt = (const float*)d_in[7];
  const float* exp_w_txt  = (const float*)d_in[8];
  const float* exp_b_txt  = (const float*)d_in[9];
  const float* mha_in_w   = (const float*)d_in[10];
  const float* mha_in_b   = (const float*)d_in[11];
  const float* mha_out_w  = (const float*)d_in[12];
  const float* mha_out_b  = (const float*)d_in[13];
  // d_in[14..21]: ln_*/mlp_* are provably dead (g1=g2=1/N exactly).
  const float* wq    = (const float*)d_in[22];
  const float* bq    = (const float*)d_in[23];
  const float* wk    = (const float*)d_in[24];
  const float* bk    = (const float*)d_in[25];
  const float* wv    = (const float*)d_in[26];
  const float* bv    = (const float*)d_in[27];
  const float* cls_w = (const float*)d_in[28];
  const float* cls_b = (const float*)d_in[29];

  const int Mi = DB * DSI;  // 4096
  const int Mt = DB * DST;  // 8192

  // Workspace layout (in floats; f16 regions counted as half-floats).
  float* w = (float*)d_ws; size_t off = 0;
  auto alloc = [&](size_t nfloats) { float* p = w + off; off += nfloats; return p; };
  float* wti = alloc((size_t)Mi * DN);
  float* wtt = alloc((size_t)Mt * DN);
  float* ie  = alloc((size_t)Mi * DH);
  float* te  = alloc((size_t)Mt * DH);
  float* qkv = alloc((size_t)Mt * 3 * DH);   // reused img then txt
  float* ao  = alloc((size_t)Mt * DH);       // per-head attn out (pre out-proj)
  float* im  = alloc((size_t)Mi * DH);
  float* tm  = alloc((size_t)Mt * DH);
  float* Qb  = alloc((size_t)Mt * DH);
  float* Kb  = alloc((size_t)Mt * DH);
  float* Vb  = alloc((size_t)Mt * DH);
  float* fi  = alloc((size_t)Mi * DH);
  float* ft  = alloc((size_t)Mt * DH);
  _Float16* xhi  = (_Float16*)alloc((size_t)Mi * DE / 2);        // f16 img_emb
  _Float16* xht  = (_Float16*)alloc((size_t)Mt * DE / 2);        // f16 txt_emb
  _Float16* ewhi = (_Float16*)alloc((size_t)DN * DH * DE / 2);   // f16 exp_w_img
  _Float16* ewht = (_Float16*)alloc((size_t)DN * DH * DE / 2);   // f16 exp_w_txt
  float* sc  = alloc((size_t)DB * DNH * DST * DST);              // score scratch

  auto mk = [](const float* A, const float* B, const float* bias, float* C,
               int M, int N, int K, i64 lda, i64 ldb, i64 ldc, float s) {
    GemmP g{}; g.A = A; g.B = B; g.bias = bias; g.C = C;
    g.M = M; g.N = N; g.K = K; g.innerDiv = 1;
    g.lda = lda; g.ldb = ldb; g.ldc = ldc; g.outScale = s;
    return g;
  };
  auto run = [&](bool transB, GemmP g, int batches) {
    dim3 grid(g.N / TILE, g.M / TILE, batches);
    if (transB) k_gemm<true><<<grid, 256, 0, stream>>>(g);
    else        k_gemm<false><<<grid, 256, 0, stream>>>(g);
  };

  // ---- one-time f16 conversion of MoE operands ----
  k_cvt_f16<<<1024, 256, 0, stream>>>(img_emb,   xhi,  (i64)Mi * DE);
  k_cvt_f16<<<1024, 256, 0, stream>>>(txt_emb,   xht,  (i64)Mt * DE);
  k_cvt_f16<<<1024, 256, 0, stream>>>(exp_w_img, ewhi, (i64)DN * DH * DE);
  k_cvt_f16<<<1024, 256, 0, stream>>>(exp_w_txt, ewht, (i64)DN * DH * DE);

  // ---- gating + fused MoE experts ----
  k_gating<<<Mi / 8, 256, 0, stream>>>(img_emb, gate_w_img, gate_b_img, wti, Mi);
  k_gating<<<Mt / 8, 256, 0, stream>>>(txt_emb, gate_w_txt, gate_b_txt, wtt, Mt);
  k_moe<<<dim3(DH / TILE, Mi / TILE), 256, 0, stream>>>(xhi, ewhi, exp_b_img, wti, ie);
  k_moe<<<dim3(DH / TILE, Mt / TILE), 256, 0, stream>>>(xht, ewht, exp_b_txt, wtt, te);

  // ---- shared-weight MHA ----
  auto runMHA = [&](int S, int M, const float* x, float* outp) {
    run(true, mk(x, mha_in_w, mha_in_b, qkv, M, 3 * DH, DH, DH, DH, 3 * DH, 1.f), 1);
    GemmP g = mk(qkv, qkv + DH, nullptr, sc, S, S, DHD, 3 * DH, 3 * DH, S, 0.125f);
    g.innerDiv = DNH;
    g.aOuter = (i64)S * 3 * DH; g.aInner = DHD;
    g.bOuter = (i64)S * 3 * DH; g.bInner = DHD;
    g.cOuter = (i64)DNH * S * S; g.cInner = (i64)S * S;
    run(true, g, DB * DNH);
    k_softmax<<<DB * DNH * S, 256, 0, stream>>>(sc, S);
    g = mk(sc, qkv + 2 * DH, nullptr, ao, S, DHD, S, S, 3 * DH, DH, 1.f);
    g.innerDiv = DNH;
    g.aOuter = (i64)DNH * S * S; g.aInner = (i64)S * S;
    g.bOuter = (i64)S * 3 * DH;  g.bInner = DHD;
    g.cOuter = (i64)S * DH;      g.cInner = DHD;
    run(false, g, DB * DNH);
    run(true, mk(ao, mha_out_w, mha_out_b, outp, M, DH, DH, DH, DH, DH, 1.f), 1);
  };
  runMHA(DSI, Mi, ie, im);
  runMHA(DST, Mt, te, tm);

  // ---- modulated attention (gates == 0.1 exactly, folded into score scale) ----
  const float modScale = 0.01f / sqrtf((float)DH);
  auto runMod = [&](int S, int M, const float* x, float* outp) {
    run(true, mk(x, wq, bq, Qb, M, DH, DH, DH, DH, DH, 1.f), 1);
    run(true, mk(x, wk, bk, Kb, M, DH, DH, DH, DH, DH, 1.f), 1);
    run(true, mk(x, wv, bv, Vb, M, DH, DH, DH, DH, DH, 1.f), 1);
    GemmP g = mk(Qb, Kb, nullptr, sc, S, S, DH, DH, DH, S, modScale);
    g.aOuter = (i64)S * DH; g.bOuter = (i64)S * DH; g.cOuter = (i64)S * S;
    run(true, g, DB);
    k_softmax<<<DB * S, 256, 0, stream>>>(sc, S);
    g = mk(sc, Vb, nullptr, outp, S, DH, S, S, DH, DH, 1.f);
    g.aOuter = (i64)S * S; g.bOuter = (i64)S * DH; g.cOuter = (i64)S * DH;
    run(false, g, DB);
  };
  runMod(DSI, Mi, im, fi);
  runMod(DST, Mt, tm, ft);

  // ---- mean-pool + classifier ----
  k_classify<<<DB, 256, 0, stream>>>(fi, ft, cls_w, cls_b, (float*)d_out);
}
